// RNN_28346784154057
// MI455X (gfx1250) — compile-verified, matrix operand order checked by reading, not run on hardware
//
#include <hip/hip_runtime.h>
#include <hip/hip_bf16.h>

typedef __bf16 bf16;
typedef __attribute__((ext_vector_type(16))) __bf16 v16bf;
typedef __attribute__((ext_vector_type(8)))  __bf16 v8bf;
typedef __attribute__((ext_vector_type(8)))  float  v8f;
typedef __attribute__((ext_vector_type(4)))  int    v4i_t;   // b128 payload type

#define RNN_B 32
#define RNN_S 2048
#define RNN_D 512
#define RNN_H 512
#define RNN_O 512

#define NWG    16     // workgroups in persistent recurrent kernel
#define COLS   32     // output columns owned per workgroup
#define LDSROW 536    // padded LDS row stride (bf16 elems): 1072B = 67*16B, conflict-free

#if defined(__has_builtin)
#  if __has_builtin(__builtin_amdgcn_global_load_async_to_lds_b128)
#    define HAVE_ASYNC_LDS 1
#  endif
#endif

#if defined(HAVE_ASYNC_LDS)
typedef __attribute__((address_space(3))) void  las_void;
typedef __attribute__((address_space(3))) v4i_t las_v4i;
__device__ __forceinline__ void wait_asynccnt0() {
#if __has_builtin(__builtin_amdgcn_s_wait_asynccnt)
    __builtin_amdgcn_s_wait_asynccnt(0);
#else
    asm volatile("s_wait_asynccnt 0x0" ::: "memory");
#endif
}
#endif

// ---------------------------------------------------------------- utilities

__global__ void init_sync_kernel(unsigned* p) {
    if (threadIdx.x < 8) p[threadIdx.x] = 0u;
}

// vectorized f32 -> bf16 (n must be a multiple of 8; all our tensors are)
__global__ void cvt_f32_bf16_kernel(const float* __restrict__ in,
                                    bf16* __restrict__ out, int n8) {
    int i = blockIdx.x * blockDim.x + threadIdx.x;
    if (i < n8) {
        const float* p = in + (size_t)i * 8;
        v8bf o;
#pragma unroll
        for (int j = 0; j < 8; ++j) o[j] = (bf16)p[j];
        *(v8bf*)(out + (size_t)i * 8) = o;
    }
}

// ------------------------------------------------- big parallel WMMA GEMM
// Cout[M,N] = A[M,K](bf16) * W[N,K]^T(bf16) + bias[N], f32 out.
// Block = 256 threads = 8 waves arranged 2(M) x 4(N); each wave computes a
// 32x64 register tile (2 M-tiles x 4 N-tiles), so a block covers 64 x 256.
// 8 WMMAs per 12 b128 fragment loads.
__global__ __launch_bounds__(256)
void wmma_gemm_bias_kernel(const bf16* __restrict__ A, const bf16* __restrict__ W,
                           const float* __restrict__ bias, float* __restrict__ Cout,
                           int M, int N, int K) {
    const int lane  = threadIdx.x & 31;
    const int wave  = threadIdx.x >> 5;   // 0..7
    const int waveM = wave >> 2;          // 0..1
    const int waveN = wave & 3;           // 0..3
    const int nBlocksN = N >> 8;          // N / 256
    const int bm = blockIdx.x / nBlocksN;
    const int bn = blockIdx.x % nBlocksN;
    const int mBase = (bm << 6) + (waveM << 5);   // 2 M-tiles of 16
    const int nBase = (bn << 8) + (waveN << 6);   // 4 N-tiles of 16
    const int l15 = lane & 15;
    const int kh  = (lane >> 4) << 3;             // per-lane K-half offset

    const bf16* pA0 = A + (size_t)(mBase + l15) * K + kh;
    const bf16* pA1 = pA0 + (size_t)16 * K;
    const bf16* pB0 = W + (size_t)(nBase + l15) * K + kh;

    v8f acc[2][4] = {};
    for (int k0 = 0; k0 < K; k0 += 32) {
        v16bf a[2], b[4];
        *(v8bf*)&a[0]       = *(const v8bf*)(pA0 + k0);
        *((v8bf*)&a[0] + 1) = *(const v8bf*)(pA0 + k0 + 16);
        *(v8bf*)&a[1]       = *(const v8bf*)(pA1 + k0);
        *((v8bf*)&a[1] + 1) = *(const v8bf*)(pA1 + k0 + 16);
#pragma unroll
        for (int ni = 0; ni < 4; ++ni) {
            const bf16* pB = pB0 + (size_t)(ni * 16) * K + k0;
            *(v8bf*)&b[ni]       = *(const v8bf*)pB;
            *((v8bf*)&b[ni] + 1) = *(const v8bf*)(pB + 16);
        }
#pragma unroll
        for (int mi = 0; mi < 2; ++mi)
#pragma unroll
            for (int ni = 0; ni < 4; ++ni)
                acc[mi][ni] = __builtin_amdgcn_wmma_f32_16x16x32_bf16(
                    false, a[mi], false, b[ni], (short)0, acc[mi][ni], false, false);
    }
    const int mc = (lane >> 4) << 3;
#pragma unroll
    for (int mi = 0; mi < 2; ++mi) {
#pragma unroll
        for (int ni = 0; ni < 4; ++ni) {
            const int   n  = nBase + ni * 16 + l15;
            const float bv = bias[n];
            float* outp = Cout + (size_t)(mBase + mi * 16 + mc) * N + n;
#pragma unroll
            for (int j = 0; j < 8; ++j)
                outp[(size_t)j * N] = acc[mi][ni][j] + bv;
        }
    }
}

// ------------------------------------------------------- grid-wide barrier
__device__ __forceinline__
void grid_barrier(unsigned* cnt, unsigned* gen, unsigned nwg, unsigned& phase) {
    __syncthreads();
    if (threadIdx.x == 0) {
        ++phase;
        __builtin_amdgcn_fence(__ATOMIC_RELEASE, "agent");   // global_wb
        unsigned arrived =
            __hip_atomic_fetch_add(cnt, 1u, __ATOMIC_RELAXED, __HIP_MEMORY_SCOPE_AGENT) + 1u;
        if (arrived == nwg * phase) {
            __hip_atomic_store(gen, phase, __ATOMIC_RELAXED, __HIP_MEMORY_SCOPE_AGENT);
        } else {
            while (__hip_atomic_load(gen, __ATOMIC_RELAXED, __HIP_MEMORY_SCOPE_AGENT) < phase)
                __builtin_amdgcn_s_sleep(2);
        }
    }
    __syncthreads();
    __builtin_amdgcn_fence(__ATOMIC_ACQUIRE, "agent");       // global_inv
}

// ----------------------------------------------- persistent recurrent kernel
// 16 workgroups x 128 threads (4 waves). Workgroup blk owns output columns
// [blk*32, blk*32+32). Weight slices live in LDS for the whole sequence.
__global__ __launch_bounds__(128)
void rnn_scan_kernel(const bf16* __restrict__ whh0, const bf16* __restrict__ wih1,
                     const bf16* __restrict__ whh1,
                     const float* __restrict__ b_ih, const float* __restrict__ b_hh,
                     const float* __restrict__ pre0,      // [B*S,H] f32 (has b_ih0)
                     bf16* __restrict__ h0buf,            // [2][B*H]
                     bf16* __restrict__ h1buf,            // [2][B*H]
                     bf16* __restrict__ h1all,            // [B*S,H]
                     unsigned* __restrict__ sync) {
    extern __shared__ char smem_raw[];
    bf16* s_whh0 = (bf16*)smem_raw;
    bf16* s_wih1 = s_whh0 + COLS * LDSROW;
    bf16* s_whh1 = s_wih1 + COLS * LDSROW;

    const int blk  = blockIdx.x;                 // 0..15
    const int c0   = blk * COLS;
    const int tid  = threadIdx.x;
    const int lane = tid & 31;
    const int wave = tid >> 5;                   // 0..3
    const int mt   = wave >> 1;                  // 0..1 (batch halves)
    const int nt   = wave & 1;                   // 0..1 (16-col sub-tile)
    const int m0   = mt << 4;
    const int nloc = (nt << 4) + (lane & 15);    // local column 0..31
    const int ncol = c0 + nloc;                  // global column
    const int kh   = (lane >> 4) << 3;
    const int mcb  = m0 + ((lane >> 4) << 3);    // C/D base row

    // ---- stage weight slices into LDS (rows = owned columns, K = 512)
#if defined(HAVE_ASYNC_LDS)
    for (int idx = tid; idx < COLS * (RNN_H / 8); idx += blockDim.x) {
        int r  = idx >> 6;                       // row 0..31
        int k8 = (idx & 63) << 3;                // element offset, 16B chunks
        __builtin_amdgcn_global_load_async_to_lds_b128(
            (v4i_t*)(whh0 + (size_t)(c0 + r) * RNN_H + k8),
            (las_v4i*)(las_void*)(s_whh0 + r * LDSROW + k8), 0, 0);
        __builtin_amdgcn_global_load_async_to_lds_b128(
            (v4i_t*)(wih1 + (size_t)(c0 + r) * RNN_H + k8),
            (las_v4i*)(las_void*)(s_wih1 + r * LDSROW + k8), 0, 0);
        __builtin_amdgcn_global_load_async_to_lds_b128(
            (v4i_t*)(whh1 + (size_t)(c0 + r) * RNN_H + k8),
            (las_v4i*)(las_void*)(s_whh1 + r * LDSROW + k8), 0, 0);
    }
    wait_asynccnt0();
#else
    for (int idx = tid; idx < COLS * (RNN_H / 8); idx += blockDim.x) {
        int r  = idx >> 6;
        int k8 = (idx & 63) << 3;
        *(v8bf*)(s_whh0 + r * LDSROW + k8) = *(const v8bf*)(whh0 + (size_t)(c0 + r) * RNN_H + k8);
        *(v8bf*)(s_wih1 + r * LDSROW + k8) = *(const v8bf*)(wih1 + (size_t)(c0 + r) * RNN_H + k8);
        *(v8bf*)(s_whh1 + r * LDSROW + k8) = *(const v8bf*)(whh1 + (size_t)(c0 + r) * RNN_H + k8);
    }
#endif
    __syncthreads();

    const bf16* pB0  = s_whh0 + nloc * LDSROW + kh;
    const bf16* pB1i = s_wih1 + nloc * LDSROW + kh;
    const bf16* pB1h = s_whh1 + nloc * LDSROW + kh;

    const float bhh0 = b_hh[ncol];                              // layer-0 b_hh
    const float bl1  = b_ih[RNN_H + ncol] + b_hh[RNN_H + ncol]; // layer-1 biases

    unsigned phase = 0;
    for (int t = 0; t < RNN_S; ++t) {
        const int   rd  = t & 1, wr = rd ^ 1;
        const bf16* h0r = h0buf + (size_t)rd * (RNN_B * RNN_H);
        bf16*       h0w = h0buf + (size_t)wr * (RNN_B * RNN_H);
        const bf16* h1r = h1buf + (size_t)rd * (RNN_B * RNN_H);
        bf16*       h1w = h1buf + (size_t)wr * (RNN_B * RNN_H);

        // ---------- layer 0: acc = pre0[t] ; acc += h0_{t-1} @ whh0^T ; tanh
        // two accumulators (even/odd K) halve the dependent-WMMA chain
        v8f acc;
        v8f accB = {};
#pragma unroll
        for (int j = 0; j < 8; ++j)
            acc[j] = pre0[((size_t)(mcb + j) * RNN_S + t) * RNN_H + ncol];
        {
            const bf16* pA = h0r + (size_t)(m0 + (lane & 15)) * RNN_H + kh;
            for (int k0 = 0; k0 < RNN_H; k0 += 64) {
                v16bf a0, b0, a1, b1;
                *(v8bf*)&a0       = *(const v8bf*)(pA + k0);
                *((v8bf*)&a0 + 1) = *(const v8bf*)(pA + k0 + 16);
                *(v8bf*)&b0       = *(const v8bf*)(pB0 + k0);
                *((v8bf*)&b0 + 1) = *(const v8bf*)(pB0 + k0 + 16);
                *(v8bf*)&a1       = *(const v8bf*)(pA + k0 + 32);
                *((v8bf*)&a1 + 1) = *(const v8bf*)(pA + k0 + 48);
                *(v8bf*)&b1       = *(const v8bf*)(pB0 + k0 + 32);
                *((v8bf*)&b1 + 1) = *(const v8bf*)(pB0 + k0 + 48);
                acc  = __builtin_amdgcn_wmma_f32_16x16x32_bf16(false, a0, false, b0,
                                                               (short)0, acc,  false, false);
                accB = __builtin_amdgcn_wmma_f32_16x16x32_bf16(false, a1, false, b1,
                                                               (short)0, accB, false, false);
            }
        }
        if (t + 1 < RNN_S) {  // warm L2/WGP$ for next step's pre0 row
            __builtin_prefetch(&pre0[((size_t)mcb * RNN_S + t + 1) * RNN_H + ncol], 0, 1);
            __builtin_prefetch(&pre0[((size_t)(mcb + 4) * RNN_S + t + 1) * RNN_H + ncol], 0, 1);
        }
#pragma unroll
        for (int j = 0; j < 8; ++j) {
            float v = tanhf(acc[j] + accB[j] + bhh0);
            h0w[(size_t)(mcb + j) * RNN_H + ncol] = (bf16)v;
        }

        grid_barrier(sync, sync + 1, NWG, phase);   // h0_t now globally visible

        // ---------- layer 1: acc = h0_t @ wih1^T + h1_{t-1} @ whh1^T + b ; tanh
        // fused K loop, independent accumulators for the ih and hh chains
        v8f acc1 = {}, acc2 = {};
        {
            const bf16* pAi = h0w + (size_t)(m0 + (lane & 15)) * RNN_H + kh;
            const bf16* pAh = h1r + (size_t)(m0 + (lane & 15)) * RNN_H + kh;
            for (int k0 = 0; k0 < RNN_H; k0 += 32) {
                v16bf ai, bi, ah, bh;
                *(v8bf*)&ai       = *(const v8bf*)(pAi + k0);
                *((v8bf*)&ai + 1) = *(const v8bf*)(pAi + k0 + 16);
                *(v8bf*)&bi       = *(const v8bf*)(pB1i + k0);
                *((v8bf*)&bi + 1) = *(const v8bf*)(pB1i + k0 + 16);
                *(v8bf*)&ah       = *(const v8bf*)(pAh + k0);
                *((v8bf*)&ah + 1) = *(const v8bf*)(pAh + k0 + 16);
                *(v8bf*)&bh       = *(const v8bf*)(pB1h + k0);
                *((v8bf*)&bh + 1) = *(const v8bf*)(pB1h + k0 + 16);
                acc1 = __builtin_amdgcn_wmma_f32_16x16x32_bf16(false, ai, false, bi,
                                                               (short)0, acc1, false, false);
                acc2 = __builtin_amdgcn_wmma_f32_16x16x32_bf16(false, ah, false, bh,
                                                               (short)0, acc2, false, false);
            }
        }
#pragma unroll
        for (int j = 0; j < 8; ++j) {
            float v  = tanhf(acc1[j] + acc2[j] + bl1);
            bf16  bv = (bf16)v;
            h1w[(size_t)(mcb + j) * RNN_H + ncol] = bv;
            h1all[((size_t)(mcb + j) * RNN_S + t) * RNN_H + ncol] = bv;
        }

        grid_barrier(sync, sync + 1, NWG, phase);   // h1_t globally visible
    }
}

// ----------------------------------------------------------------- launcher

extern "C" void kernel_launch(void* const* d_in, const int* in_sizes, int n_in,
                              void* d_out, int out_size, void* d_ws, size_t ws_size,
                              hipStream_t stream) {
    const float* x     = (const float*)d_in[0];   // [B,S,D]
    const float* hid   = (const float*)d_in[1];   // [L,B,H]
    const float* w_ih  = (const float*)d_in[2];   // [L,H,D]
    const float* w_hh  = (const float*)d_in[3];   // [L,H,H]
    const float* b_ih  = (const float*)d_in[4];   // [L,H]
    const float* b_hh  = (const float*)d_in[5];   // [L,H]
    const float* w_out = (const float*)d_in[6];   // [O,H]
    const float* b_out = (const float*)d_in[7];   // [O]
    float*       out   = (float*)d_out;           // [B,S,O]

    const size_t BS   = (size_t)RNN_B * RNN_S;
    const size_t WMAT = (size_t)RNN_H * RNN_D;    // 512*512

    char*  ws  = (char*)d_ws;
    size_t off = 0;
    auto alloc = [&](size_t bytes) -> void* {
        void* p = ws + off;
        off += (bytes + 255) & ~(size_t)255;
        return p;
    };

    unsigned* syncp  = (unsigned*)alloc(256);
    bf16*     xb     = (bf16*)alloc(BS * RNN_D * 2);
    bf16*     wih0b  = (bf16*)alloc(WMAT * 2);
    bf16*     wih1b  = (bf16*)alloc(WMAT * 2);
    bf16*     whh0b  = (bf16*)alloc(WMAT * 2);
    bf16*     whh1b  = (bf16*)alloc(WMAT * 2);
    bf16*     woutb  = (bf16*)alloc(WMAT * 2);
    bf16*     h0buf  = (bf16*)alloc(2 * (size_t)RNN_B * RNN_H * 2);
    bf16*     h1buf  = (bf16*)alloc(2 * (size_t)RNN_B * RNN_H * 2);
    float*    pre0   = (float*)alloc(BS * RNN_H * 4);
    bf16*     h1all  = (bf16*)alloc(BS * RNN_H * 2);

    init_sync_kernel<<<1, 64, 0, stream>>>(syncp);

    auto cvt = [&](const float* src, bf16* dst, size_t n) {
        int n8 = (int)(n / 8);
        cvt_f32_bf16_kernel<<<(n8 + 255) / 256, 256, 0, stream>>>(src, dst, n8);
    };
    cvt(x, xb, BS * RNN_D);
    cvt(w_ih,        wih0b, WMAT);
    cvt(w_ih + WMAT, wih1b, WMAT);
    cvt(w_hh,        whh0b, WMAT);
    cvt(w_hh + WMAT, whh1b, WMAT);
    cvt(w_out,       woutb, WMAT);
    cvt(hid,                         h0buf, (size_t)RNN_B * RNN_H);  // layer0 h @ slot 0
    cvt(hid + (size_t)RNN_B * RNN_H, h1buf, (size_t)RNN_B * RNN_H);  // layer1 h @ slot 0

    // pre0 = x @ w_ih0^T + b_ih0   (M = B*S, N = H, K = D)
    {
        int grid = (int)((BS / 64) * (RNN_H / 256));
        wmma_gemm_bias_kernel<<<grid, 256, 0, stream>>>(xb, wih0b, b_ih, pre0,
                                                        (int)BS, RNN_H, RNN_D);
    }

    // persistent recurrent scan
    {
        size_t lds = (size_t)3 * COLS * LDSROW * sizeof(bf16);   // ~100.5 KB
        rnn_scan_kernel<<<NWG, 128, lds, stream>>>(whh0b, wih1b, whh1b,
                                                   b_ih, b_hh, pre0,
                                                   h0buf, h1buf, h1all, syncp);
    }

    // out = h1all @ w_out^T + b_out   (M = B*S, N = O, K = H)
    {
        int grid = (int)((BS / 64) * (RNN_O / 256));
        wmma_gemm_bias_kernel<<<grid, 256, 0, stream>>>(h1all, woutb, b_out, out,
                                                        (int)BS, RNN_O, RNN_H);
    }
}
